// Attention_47974784696998
// MI455X (gfx1250) — matrix operand order repaired and checked by reading, not compiled, not run on hardware
//
#include <hip/hip_runtime.h>
#include <stdint.h>

// ---------------------------------------------------------------------------
// Problem sizes (fixed by the reference): B=64, S=1024, H=E=A=1024
// ---------------------------------------------------------------------------
#define BB 64
#define SS 1024
#define EE 1024
#define AA 1024

typedef __attribute__((ext_vector_type(16))) __bf16 v16bf;
typedef __attribute__((ext_vector_type(8)))  float  v8f;

union FragU {
    uint32_t d[8];
    uint4    q[2];
    v16bf    v;
};

__device__ __forceinline__ uint16_t f2bf(float f) {
    uint32_t u = __float_as_uint(f);
    uint32_t r = u + 0x7FFFu + ((u >> 16) & 1u);   // round-to-nearest-even
    return (uint16_t)(r >> 16);
}

// Branch-free tanh: 1 - 2/(1 + e^{2x}).  v_exp_f32 + v_rcp_f32, no exec-mask
// branching (the libm tanhf expansion costs ~40 branchy VALU/SALU ops per
// element right after the WMMA hazard window).
__device__ __forceinline__ float fast_tanh(float x) {
    float e = __expf(2.0f * x);                       // inf for large x: OK
    return 1.0f - 2.0f * __builtin_amdgcn_rcpf(1.0f + e);
}

// ---------------------------------------------------------------------------
// Kernel 1: repack U_a (fp32 [E][A], row-major) into bf16 WMMA B-fragment
// layout.  Packed index: (((nt*32 + kt)*32 + lane)*16 + v*2 + j)
// B 32x16 layout used here (contiguous K per lane half, not ISA table order): lane holds column n = lane&15, half = lane>>4,
// K = kt*32 + half*16 + v*2 + j   (VGPR v holds K pair {2v,2v+1} per half)
// ---------------------------------------------------------------------------
__global__ void pack_Ua_kernel(const float* __restrict__ Ua,
                               uint16_t* __restrict__ pk) {
    uint32_t idx  = blockIdx.x * 256u + threadIdx.x;   // 2^20 total
    uint32_t e    = idx & 15u;
    uint32_t lane = (idx >> 4) & 31u;
    uint32_t kt   = (idx >> 9) & 31u;
    uint32_t nt   = idx >> 14;
    uint32_t v    = e >> 1, j = e & 1u;
    uint32_t k    = kt * 32u + (lane >> 4) * 16u + v * 2u + j;
    uint32_t n    = nt * 16u + (lane & 15u);
    pk[idx] = f2bf(Ua[(size_t)k * AA + n]);
}

// ---------------------------------------------------------------------------
// Kernel 2: kdec[b][a] = sum_h dec[b][h] * W_a[h][a]   (fp32, tiny GEMM)
// ---------------------------------------------------------------------------
__global__ void kdec_kernel(const float* __restrict__ dec,
                            const float* __restrict__ Wa,
                            float* __restrict__ kdec) {
    int b = blockIdx.x >> 2;
    int a = (blockIdx.x & 3) * 256 + threadIdx.x;
    __shared__ float sd[256];
    float acc = 0.f;
    for (int h0 = 0; h0 < 1024; h0 += 256) {
        __syncthreads();
        sd[threadIdx.x] = dec[b * 1024 + h0 + threadIdx.x];
        __syncthreads();
        #pragma unroll 8
        for (int i = 0; i < 256; ++i)
            acc += sd[i] * Wa[(size_t)(h0 + i) * AA + a];
    }
    kdec[b * AA + a] = acc;
}

// ---------------------------------------------------------------------------
// Kernel 3 (hot): fused  scores[b][s] = sum_a v_a[a]*tanh(kdec[b][a] +
//                 (enc[b][s] @ U_a)[a])  via bf16 WMMA, never materializing
//                 k_enc.  One workgroup = (b, 16-row s-tile); 4 waves split
//                 the 64 a-tiles; K loop over E in steps of 32.
// ---------------------------------------------------------------------------
#define LDS_STRIDE 1032   // 1024 + 8 bf16 pad -> conflict-free ds_load_b128

__global__ __launch_bounds__(128)
void scores_kernel(const float* __restrict__ enc,
                   const uint16_t* __restrict__ pk,
                   const float* __restrict__ kdec,
                   const float* __restrict__ va,
                   float* __restrict__ scores) {
    __shared__ uint16_t smA[16 * LDS_STRIDE];   // 33 KB: 16 x 1024 bf16 tile
    __shared__ float    ssc[16];

    int tid = threadIdx.x;
    int b   = blockIdx.x >> 6;
    int s0  = (blockIdx.x & 63) * 16;

    if (tid < 16) ssc[tid] = 0.f;

    // ---- stage encoder tile -> LDS (fp32 -> bf16, padded row stride) ----
    const float* encT = enc + ((size_t)(b * SS + s0)) * EE;
    for (int p = tid; p < 16 * 512; p += 128) {
        int row = p >> 9;
        int c2  = (p & 511) * 2;
        float2 f = *reinterpret_cast<const float2*>(encT + (size_t)row * EE + c2);
        uint32_t packed = (uint32_t)f2bf(f.x) | ((uint32_t)f2bf(f.y) << 16);
        *reinterpret_cast<uint32_t*>(&smA[row * LDS_STRIDE + c2]) = packed;
    }
    __syncthreads();

    int lane  = tid & 31;
    int w     = tid >> 5;          // wave id, 0..3
    int half  = lane >> 4;         // 0 or 1
    int nlo   = lane & 15;
    int rowOff = nlo * LDS_STRIDE; // A-frag: lane&15 selects matrix row M
    int half8  = half * 16;        // A-frag: K offset for lane half

    float sr[8];
    #pragma unroll
    for (int r = 0; r < 8; ++r) sr[r] = 0.f;

    for (int nt = w; nt < 64; nt += 4) {
        int   n   = nt * 16 + nlo;
        float vva = va[n];
        float kd  = kdec[b * AA + n];

        v8f acc = {0.f, 0.f, 0.f, 0.f, 0.f, 0.f, 0.f, 0.f};
        const uint16_t* bbase = pk + (size_t)nt * (32 * 512) + lane * 16;

        #pragma unroll 4
        for (int kt = 0; kt < 32; ++kt) {
            // A fragment (16x32 bf16): lanes 0-15 K=0..15+base,
            // lanes 16-31 K=16..31+base -> two contiguous 16B LDS loads.
            const uint16_t* aptr = &smA[rowOff + kt * 32 + half8];
            FragU fa, fb;
            fa.q[0] = *reinterpret_cast<const uint4*>(aptr);
            fa.q[1] = *reinterpret_cast<const uint4*>(aptr + 8);
            // B fragment (32x16 bf16): packed contiguous 32B per lane.
            const uint16_t* bptr = bbase + kt * 512;
            fb.q[0] = *reinterpret_cast<const uint4*>(bptr);
            fb.q[1] = *reinterpret_cast<const uint4*>(bptr + 8);
            __builtin_prefetch(bptr + 512, 0, 1);   // next-kt B fragment
            acc = __builtin_amdgcn_wmma_f32_16x16x32_bf16(
                false, fa.v, false, fb.v, (short)0, acc, false, false);
        }
        // Epilogue: C layout -> lane holds column n for rows m = 8*half + r.
        #pragma unroll
        for (int r = 0; r < 8; ++r)
            sr[r] += vva * fast_tanh(acc[r] + kd);
    }

    // Reduce over the 16 columns held within each lane-half.
    #pragma unroll
    for (int m = 1; m < 16; m <<= 1)
        #pragma unroll
        for (int r = 0; r < 8; ++r)
            sr[r] += __shfl_xor(sr[r], m, 32);

    if (nlo == 0) {
        #pragma unroll
        for (int r = 0; r < 8; ++r)
            atomicAdd(&ssc[half * 8 + r], sr[r]);   // ds_add_f32 across waves
    }
    __syncthreads();
    if (tid < 16)
        scores[(size_t)b * SS + s0 + tid] = ssc[tid];
}

// ---------------------------------------------------------------------------
// Kernel 4: masked softmax over S per batch row.
// ---------------------------------------------------------------------------
__global__ void softmax_kernel(const float* __restrict__ scores,
                               const uint8_t* __restrict__ mask,
                               float* __restrict__ weights) {
    int b = blockIdx.x, tid = threadIdx.x;
    __shared__ float red[256];

    float v[4];
    #pragma unroll
    for (int i = 0; i < 4; ++i) {
        int s = tid + i * 256;
        float sc = scores[b * SS + s];
        if (!mask[b * SS + s]) sc += -1e9f;
        v[i] = sc;
    }
    float m = fmaxf(fmaxf(v[0], v[1]), fmaxf(v[2], v[3]));
    red[tid] = m; __syncthreads();
    for (int o = 128; o > 0; o >>= 1) {
        if (tid < o) red[tid] = fmaxf(red[tid], red[tid + o]);
        __syncthreads();
    }
    m = red[0]; __syncthreads();

    float e[4], s = 0.f;
    #pragma unroll
    for (int i = 0; i < 4; ++i) { e[i] = __expf(v[i] - m); s += e[i]; }
    red[tid] = s; __syncthreads();
    for (int o = 128; o > 0; o >>= 1) {
        if (tid < o) red[tid] += red[tid + o];
        __syncthreads();
    }
    float inv = 1.f / red[0];
    #pragma unroll
    for (int i = 0; i < 4; ++i)
        weights[b * SS + tid + i * 256] = e[i] * inv;
}

// ---------------------------------------------------------------------------
// Kernel 5: ctx[b][e] = sum_s weights[b][s] * enc[b][s][e]  (memory bound)
// ---------------------------------------------------------------------------
__global__ void ctx_kernel(const float* __restrict__ weights,
                           const float* __restrict__ enc,
                           float* __restrict__ ctx) {
    int b = blockIdx.x >> 2;
    int e = (blockIdx.x & 3) * 256 + threadIdx.x;
    __shared__ float sw[256];
    float acc = 0.f;
    for (int sBase = 0; sBase < SS; sBase += 256) {
        __syncthreads();
        sw[threadIdx.x] = weights[b * SS + sBase + threadIdx.x];
        __syncthreads();
        #pragma unroll 4
        for (int i = 0; i < 256; ++i)
            acc += sw[i] * enc[((size_t)(b * SS + sBase + i)) * EE + e];
    }
    ctx[b * EE + e] = acc;
}

// ---------------------------------------------------------------------------
// Host launcher
// ---------------------------------------------------------------------------
extern "C" void kernel_launch(void* const* d_in, const int* in_sizes, int n_in,
                              void* d_out, int out_size, void* d_ws, size_t ws_size,
                              hipStream_t stream) {
    const float*   dec  = (const float*)d_in[0];   // [B,H]
    const float*   enc  = (const float*)d_in[1];   // [B,S,E]
    const uint8_t* mask = (const uint8_t*)d_in[2]; // [B,S] numpy bool
    const float*   Wa   = (const float*)d_in[3];   // [H,A]
    const float*   Ua   = (const float*)d_in[4];   // [E,A]
    const float*   va   = (const float*)d_in[5];   // [A]

    float* out     = (float*)d_out;
    float* ctx     = out;                 // [B,E]
    float* weights = out + BB * EE;       // [B,S]

    // Workspace layout: 2 MB packed bf16 U_a | 256 KB kdec | 256 KB scores
    uint16_t* pk     = (uint16_t*)d_ws;
    float*    kdec   = (float*)((char*)d_ws + (2u << 20));
    float*    scores = kdec + BB * AA;

    pack_Ua_kernel<<<4096, 256, 0, stream>>>(Ua, pk);
    kdec_kernel  <<<BB * 4, 256, 0, stream>>>(dec, Wa, kdec);
    scores_kernel<<<BB * (SS / 16), 128, 0, stream>>>(enc, pk, kdec, va, scores);
    softmax_kernel<<<BB, 256, 0, stream>>>(scores, mask, weights);
    ctx_kernel   <<<BB * 4, 256, 0, stream>>>(weights, enc, ctx);
}
